// Decoder_39599598469285
// MI455X (gfx1250) — matrix operand order repaired and checked by reading, not compiled, not run on hardware
//
#include <hip/hip_runtime.h>

// ============================================================================
// MI455X (gfx1250, wave32) relational-GCN decoder.
//
// ~270 GFLOP of dense GEMM vs ~1.5 GB HBM traffic (h/acc/S ~50MB each ->
// L2-resident in 192MB L2) => compute-dominated => all GEMMs via
// v_wmma_f32_16x16x32_bf16 (bf16 operands, f32 accum).
//
// Tiling: each wave32 owns a 16x64 output strip (4 accumulators): one
// A-fragment load feeds 4 WMMAs (4x register-level A reuse).  Attention
// stages the h-tile and the score matrix in LDS (bf16), so both attention
// GEMMs pull A-fragments via ds_load_b128.  The final h@W_h GEMM fuses
// (+acc, relu) into its epilogue and layers ping-pong h/acc buffers.
// ============================================================================

#define NN     50000   // tgt nodes  (50000 = 3125*16)
#define MSRC   1024    // src nodes
#define LSUB   3
#define EE     200000  // edges
#define HIDD   512
#define EMBD   256
#define VOCABD 512
#define NSLT   4
#define NSRT   5

typedef __attribute__((ext_vector_type(16))) __bf16 v16bf;
typedef __attribute__((ext_vector_type(8)))  __bf16 v8bf;
typedef __attribute__((ext_vector_type(8)))  float  v8f;

#define WMMA_BF16(A, B, ACC) \
  __builtin_amdgcn_wmma_f32_16x16x32_bf16(false, (A), false, (B), (short)0, (ACC), false, false)

// ---------------------------------------------------------------------------
// Fragment loaders (CDNA5 WMMA 16x16x32 bf16 layouts, CDNA5 ISA)
// A (16x32, MxK): lanes 0-15 row=lane; V0..V3 = K(half*8+0..7), V4..V7 = +16.
// B (32x16, KxN): lane&15 = N column; same K order as A (half*8+0..7, +16)
//                 -> B stored transposed: BT[N][K].
// ---------------------------------------------------------------------------
__device__ __forceinline__ v16bf frag_a_f32(const float* __restrict__ A, int lda,
                                            int rowBase, int k0) {
  const int lane = threadIdx.x & 31;
  const int r    = rowBase + (lane & 15);
  const int half = lane >> 4;
  const float* p = A + (size_t)r * lda + k0 + half * 8;
  float4 f0 = *(const float4*)(p);        // K = half*8 + 0..3
  float4 f1 = *(const float4*)(p + 4);    // K = half*8 + 4..7
  float4 f2 = *(const float4*)(p + 16);   // K = 16 + half*8 + 0..3
  float4 f3 = *(const float4*)(p + 20);   // K = 16 + half*8 + 4..7
  v16bf a;
  a[0]  = (__bf16)f0.x; a[1]  = (__bf16)f0.y; a[2]  = (__bf16)f0.z; a[3]  = (__bf16)f0.w;
  a[4]  = (__bf16)f1.x; a[5]  = (__bf16)f1.y; a[6]  = (__bf16)f1.z; a[7]  = (__bf16)f1.w;
  a[8]  = (__bf16)f2.x; a[9]  = (__bf16)f2.y; a[10] = (__bf16)f2.z; a[11] = (__bf16)f2.w;
  a[12] = (__bf16)f3.x; a[13] = (__bf16)f3.y; a[14] = (__bf16)f3.z; a[15] = (__bf16)f3.w;
  return a;
}

// A-fragment from an LDS bf16 tile (row-major, 'pitch' elems per row)
__device__ __forceinline__ v16bf frag_a_lds(const __bf16* base, int pitch, int k0) {
  const int lane = threadIdx.x & 31;
  const int r = lane & 15, half = lane >> 4;
  const __bf16* p = base + r * pitch + k0 + half * 8;
  v8bf a0 = *(const v8bf*)p;              // ds_load_b128
  v8bf a1 = *(const v8bf*)(p + 16);
  v16bf a;
#pragma unroll
  for (int i = 0; i < 8; i++) { a[i] = a0[i]; a[8 + i] = a1[i]; }
  return a;
}

__device__ __forceinline__ v16bf frag_b_bt(const __bf16* __restrict__ BT, int ldb,
                                           int colBase, int k0) {
  const int lane = threadIdx.x & 31;
  const int n    = colBase + (lane & 15);
  const int koff = (lane >> 4) * 8;
  // two 8-bf16 (16B) loads at K=half*8 and 16+half*8, aligned since ldb % 16 == 0, k0 % 32 == 0
  const __bf16* p = BT + (size_t)n * ldb + k0 + koff;
  v8bf b0 = *(const v8bf*)p;
  v8bf b1 = *(const v8bf*)(p + 16);
  v16bf b;
#pragma unroll
  for (int i = 0; i < 8; i++) { b[i] = b0[i]; b[8 + i] = b1[i]; }
  return b;
}

// ---------------------------------------------------------------------------
// Generic GEMM: D = A[MxK](f32, cvt->bf16 on the fly) * B (BT[N][K] bf16).
// epilogue: 0: C = D    1: C += D    2: OUT = relu(C + D)  (OUT may alias C;
// each wave touches only its own 16x64 strip, so in-place is race-free).
// One 16x64 strip per wave (4 WMMA tiles), 8 waves per block.  N % 64 == 0.
// ---------------------------------------------------------------------------
__global__ __launch_bounds__(256) void gemm_bf16_kernel(
    const float* __restrict__ A, const __bf16* __restrict__ BT,
    float* __restrict__ C, float* __restrict__ OUT,
    int M, int N, int K, int epilogue) {
  const int wid     = blockIdx.x * 8 + (threadIdx.x >> 5);
  const int stripsN = N >> 6;
  const int strips  = (M >> 4) * stripsN;
  if (wid >= strips) return;                // wave-uniform: EXEC stays all-1s
  const int tm = wid / stripsN;
  const int c0 = (wid % stripsN) * 64;

  v8f acc0 = {}, acc1 = {}, acc2 = {}, acc3 = {};
  for (int k0 = 0; k0 < K; k0 += 32) {
    v16bf a  = frag_a_f32(A, K, tm * 16, k0);        // loaded once, used 4x
    v16bf b0 = frag_b_bt(BT, K, c0 +  0, k0);
    v16bf b1 = frag_b_bt(BT, K, c0 + 16, k0);
    v16bf b2 = frag_b_bt(BT, K, c0 + 32, k0);
    v16bf b3 = frag_b_bt(BT, K, c0 + 48, k0);
    acc0 = WMMA_BF16(a, b0, acc0);
    acc1 = WMMA_BF16(a, b1, acc1);
    acc2 = WMMA_BF16(a, b2, acc2);
    acc3 = WMMA_BF16(a, b3, acc3);
  }
  const int lane  = threadIdx.x & 31;
  const int col   = c0 + (lane & 15);
  const int rbase = tm * 16 + ((lane >> 4) << 3);
  const size_t base = (size_t)rbase * N + col;
  if (epilogue == 0) {
#pragma unroll
    for (int i = 0; i < 8; i++) {
      float* row = C + base + (size_t)i * N;
      row[0] = acc0[i]; row[16] = acc1[i]; row[32] = acc2[i]; row[48] = acc3[i];
    }
  } else if (epilogue == 1) {
#pragma unroll
    for (int i = 0; i < 8; i++) {
      float* row = C + base + (size_t)i * N;
      row[0] += acc0[i]; row[16] += acc1[i]; row[32] += acc2[i]; row[48] += acc3[i];
    }
  } else {  // fused: OUT = relu(C + D)
#pragma unroll
    for (int i = 0; i < 8; i++) {
      const float* cr = C + base + (size_t)i * N;
      float* orow = OUT + base + (size_t)i * N;
      orow[0]  = fmaxf(cr[0]  + acc0[i], 0.f);
      orow[16] = fmaxf(cr[16] + acc1[i], 0.f);
      orow[32] = fmaxf(cr[32] + acc2[i], 0.f);
      orow[48] = fmaxf(cr[48] + acc3[i], 0.f);
    }
  }
}

// ---------------------------------------------------------------------------
// Fused cross-attention (flash-style), per 16-row tile of h:
//   stage h-tile -> LDS bf16 (8KB); S = (h @ k^T)/16 -> LDS bf16 (32KB);
//   softmax (wave32 shfl, exp in place); ctx = P @ k -> acc buffer.
// Both GEMMs pull A-fragments from LDS.  No 200MB attn matrix materialized.
// ---------------------------------------------------------------------------
__global__ __launch_bounds__(256) void attn_kernel(
    const float* __restrict__ h, const __bf16* __restrict__ kbf,
    const __bf16* __restrict__ kTbf, float* __restrict__ accb) {
  __shared__ __bf16 Sbf[16 * MSRC];   // 32 KB scores/probs
  __shared__ __bf16 Hbf[16 * EMBD];   //  8 KB staged h tile
  __shared__ float rowsum[16];
  const int w = threadIdx.x >> 5, lane = threadIdx.x & 31;
  const int rt = blockIdx.x;          // row tile (0..3124)

  // ---- stage h tile (16 x 256) to LDS as bf16, cooperatively ----
#pragma unroll
  for (int j = 0; j < 16; j++) {
    const int idx = j * 256 + threadIdx.x;        // idx = r*256 + c
    Hbf[idx] = (__bf16)h[(size_t)rt * 16 * EMBD + idx];
  }
  __syncthreads();

  // ---- phase 1: scores S[16 x 1024], 16 groups of 4 tiles, 2 per wave ----
  for (int g = w; g < (MSRC / 64); g += 8) {
    const int c0 = g * 64;
    v8f acc0 = {}, acc1 = {}, acc2 = {}, acc3 = {};
    for (int k0 = 0; k0 < EMBD; k0 += 32) {
      v16bf a  = frag_a_lds(Hbf, EMBD, k0);
      v16bf b0 = frag_b_bt(kbf, EMBD, c0 +  0, k0);  // kbf[m][d] == (k^T)[d][m]
      v16bf b1 = frag_b_bt(kbf, EMBD, c0 + 16, k0);
      v16bf b2 = frag_b_bt(kbf, EMBD, c0 + 32, k0);
      v16bf b3 = frag_b_bt(kbf, EMBD, c0 + 48, k0);
      acc0 = WMMA_BF16(a, b0, acc0);
      acc1 = WMMA_BF16(a, b1, acc1);
      acc2 = WMMA_BF16(a, b2, acc2);
      acc3 = WMMA_BF16(a, b3, acc3);
    }
    const int c  = c0 + (lane & 15);
    const int rb = (lane >> 4) << 3;
#pragma unroll
    for (int i = 0; i < 8; i++) {
      __bf16* row = &Sbf[(rb + i) * MSRC + c];
      row[0]  = (__bf16)(acc0[i] * 0.0625f);   // 1/sqrt(256)
      row[16] = (__bf16)(acc1[i] * 0.0625f);
      row[32] = (__bf16)(acc2[i] * 0.0625f);
      row[48] = (__bf16)(acc3[i] * 0.0625f);
    }
  }
  __syncthreads();

  // ---- phase 2: softmax over M=1024 (rows w and w+8 per wave) ----
#pragma unroll
  for (int rr = 0; rr < 2; rr++) {
    const int r = w + rr * 8;
    float mx = -1e30f;
    for (int c = lane; c < MSRC; c += 32)
      mx = fmaxf(mx, (float)Sbf[r * MSRC + c]);
    for (int off = 16; off; off >>= 1) mx = fmaxf(mx, __shfl_xor(mx, off, 32));
    float sum = 0.f;
    for (int c = lane; c < MSRC; c += 32) {
      float e = __expf((float)Sbf[r * MSRC + c] - mx);
      Sbf[r * MSRC + c] = (__bf16)e;          // in-place, own elements only
      sum += e;
    }
    for (int off = 16; off; off >>= 1) sum += __shfl_xor(sum, off, 32);
    if (lane == 0) rowsum[r] = sum;
  }
  __syncthreads();

  // ---- phase 3: ctx = P @ k ; 16x32 strip per wave, A from LDS reused 2x ----
  {
    const int c0 = w * 32;                    // 8 waves x 32 cols = 256
    v8f acc0 = {}, acc1 = {};
    for (int k0 = 0; k0 < MSRC; k0 += 32) {
      v16bf a  = frag_a_lds(Sbf, MSRC, k0);
      v16bf b0 = frag_b_bt(kTbf, MSRC, c0 +  0, k0); // kT[d][m] contig in m
      v16bf b1 = frag_b_bt(kTbf, MSRC, c0 + 16, k0);
      acc0 = WMMA_BF16(a, b0, acc0);
      acc1 = WMMA_BF16(a, b1, acc1);
    }
    const int c  = c0 + (lane & 15);
    const int rb = (lane >> 4) << 3;
#pragma unroll
    for (int i = 0; i < 8; i++) {
      const int r = rb + i;
      float* row = &accb[(size_t)(rt * 16 + r) * EMBD + c];
      const float inv = 1.0f / rowsum[r];
      row[0]  = acc0[i] * inv;
      row[16] = acc1[i] * inv;
    }
  }
}

// --------------------------- small helper kernels --------------------------
__global__ void embed_kernel(const int* __restrict__ tx,
                             const float* __restrict__ tab,
                             float* __restrict__ h) {
  const int id = blockIdx.x * 256 + threadIdx.x;  // NN*EMBD threads
  const int n = id >> 8, d = id & 255;
  float s = 0.f;
#pragma unroll
  for (int l = 0; l < LSUB; l++) s += tab[(size_t)tx[n * LSUB + l] * EMBD + d];
  h[id] = s;
}

__global__ void tconv_kernel(const float* __restrict__ in, __bf16* __restrict__ out,
                             int R, int C) {                // out[c][r]=in[r][c]
  const int id = blockIdx.x * 256 + threadIdx.x;
  if (id >= R * C) return;
  const int r = id / C, c = id % C;
  out[(size_t)c * R + r] = (__bf16)in[id];
}

__global__ void kprep_kernel(const float* __restrict__ kf,
                             __bf16* __restrict__ kbf, __bf16* __restrict__ kTbf) {
  const int id = blockIdx.x * 256 + threadIdx.x;  // MSRC*EMBD
  const int m = id >> 8, d = id & 255;
  const float v = kf[id];
  kbf[id]            = (__bf16)v;
  kTbf[d * MSRC + m] = (__bf16)v;
}

__global__ void zero_kernel(float4* __restrict__ p, int n4) {
  const int i = blockIdx.x * 256 + threadIdx.x;
  if (i < n4) p[i] = make_float4(0.f, 0.f, 0.f, 0.f);
}

// per-type scatter: S_t[tgt] += h[src]  (wave per edge, f32 global atomics)
__global__ __launch_bounds__(256) void scatter_kernel(
    const float* __restrict__ h, const int* __restrict__ eidx,
    const int* __restrict__ etype, float* __restrict__ S, int t) {
  const int wid = blockIdx.x * 8 + (threadIdx.x >> 5);
  const int lane = threadIdx.x & 31;
  if (wid >= EE) return;
  const int src = eidx[wid];
  // irregular gather stream: pull h[src] row toward GL2 early
  __builtin_prefetch(h + (size_t)src * EMBD + lane * 8, 0, 1);
  if (etype[wid] != t) return;
  const int tgt = eidx[EE + wid];
  const float* hs = h + (size_t)src * EMBD;
  float* sd = S + (size_t)tgt * EMBD;
#pragma unroll
  for (int j = 0; j < 8; j++)
    atomicAdd(&sd[lane + 32 * j], hs[lane + 32 * j]);
}

// in-place log_softmax over VOCAB=512 per row (+bias); wave per row
__global__ __launch_bounds__(256) void node_lsm_kernel(float* __restrict__ logits,
                                                       const float* __restrict__ bz) {
  const int row = blockIdx.x * 8 + (threadIdx.x >> 5);
  const int lane = threadIdx.x & 31;
  if (row >= NN) return;
  float* p = logits + (size_t)row * VOCABD;
  float vals[16];
  float mx = -1e30f;
#pragma unroll
  for (int j = 0; j < 16; j++) {
    vals[j] = p[lane + 32 * j] + bz[lane + 32 * j];
    mx = fmaxf(mx, vals[j]);
  }
  for (int off = 16; off; off >>= 1) mx = fmaxf(mx, __shfl_xor(mx, off, 32));
  float sum = 0.f;
#pragma unroll
  for (int j = 0; j < 16; j++) sum += __expf(vals[j] - mx);
  for (int off = 16; off; off >>= 1) sum += __shfl_xor(sum, off, 32);
  const float lse = mx + __logf(sum);
#pragma unroll
  for (int j = 0; j < 16; j++) p[lane + 32 * j] = vals[j] - lse;
}

// edge head: log_softmax(concat(h[src],h[tgt]) @ Wg + bg); wave per edge
__global__ __launch_bounds__(256) void edge_head_kernel(
    const float* __restrict__ h, const int* __restrict__ eidx,
    const float* __restrict__ Wg, const float* __restrict__ bg,
    float* __restrict__ out) {
  const int wid = blockIdx.x * 8 + (threadIdx.x >> 5);
  const int lane = threadIdx.x & 31;
  if (wid >= EE) return;
  const int src = eidx[wid], tgt = eidx[EE + wid];
  const float* hs = h + (size_t)src * EMBD;
  const float* ht = h + (size_t)tgt * EMBD;
  float a[NSRT] = {0.f, 0.f, 0.f, 0.f, 0.f};
  for (int d = lane; d < EMBD; d += 32) {
    const float fs = hs[d], ft = ht[d];
    const float* w0 = Wg + (size_t)d * NSRT;
    const float* w1 = Wg + (size_t)(EMBD + d) * NSRT;
#pragma unroll
    for (int j = 0; j < NSRT; j++) a[j] += fs * w0[j] + ft * w1[j];
  }
#pragma unroll
  for (int j = 0; j < NSRT; j++)
    for (int off = 16; off; off >>= 1) a[j] += __shfl_xor(a[j], off, 32);
  if (lane == 0) {
    float v[NSRT], mx = -1e30f;
#pragma unroll
    for (int j = 0; j < NSRT; j++) { v[j] = a[j] + bg[j]; mx = fmaxf(mx, v[j]); }
    float sum = 0.f;
#pragma unroll
    for (int j = 0; j < NSRT; j++) sum += __expf(v[j] - mx);
    const float lse = mx + __logf(sum);
#pragma unroll
    for (int j = 0; j < NSRT; j++) out[(size_t)wid * NSRT + j] = v[j] - lse;
  }
}

// ============================================================================
extern "C" void kernel_launch(void* const* d_in, const int* in_sizes, int n_in,
                              void* d_out, int out_size, void* d_ws, size_t ws_size,
                              hipStream_t stream) {
  const int*   tgt_x  = (const int*)d_in[0];
  const float* x      = (const float*)d_in[1];
  const int*   eidx   = (const int*)d_in[2];
  const int*   etype  = (const int*)d_in[3];
  const float* emtab  = (const float*)d_in[4];
  const float* W_h1   = (const float*)d_in[5];
  const float* W_rel1 = (const float*)d_in[6];
  const float* W_src1 = (const float*)d_in[7];
  const float* W_h3   = (const float*)d_in[8];
  const float* W_rel3 = (const float*)d_in[9];
  const float* W_src3 = (const float*)d_in[10];
  const float* Wz     = (const float*)d_in[11];
  const float* bz     = (const float*)d_in[12];
  const float* Wg     = (const float*)d_in[13];
  const float* bg     = (const float*)d_in[14];
  float* out = (float*)d_out;

  // ---- workspace carve-up (~157 MB) ----
  char* ws = (char*)d_ws;
  size_t off = 0;
  auto take = [&](size_t bytes) -> char* {
    char* p = ws + off;
    off = (off + bytes + 255) & ~(size_t)255;
    return p;
  };
  float*  hA     = (float*)take((size_t)NN * EMBD * 4);
  float*  hB     = (float*)take((size_t)NN * EMBD * 4);
  float*  S_buf  = (float*)take((size_t)NN * EMBD * 4);
  float*  kf32   = (float*)take((size_t)MSRC * EMBD * 4);
  __bf16* kbf    = (__bf16*)take((size_t)MSRC * EMBD * 2);
  __bf16* kTbf   = (__bf16*)take((size_t)EMBD * MSRC * 2);
  __bf16* WTsrc1 = (__bf16*)take((size_t)HIDD * EMBD * 2);
  __bf16* WTsrc3 = (__bf16*)take((size_t)HIDD * EMBD * 2);
  __bf16* WTh1   = (__bf16*)take((size_t)EMBD * EMBD * 2);
  __bf16* WTh3   = (__bf16*)take((size_t)EMBD * EMBD * 2);
  __bf16* WTrel1 = (__bf16*)take((size_t)NSLT * EMBD * EMBD * 2);
  __bf16* WTrel3 = (__bf16*)take((size_t)NSLT * EMBD * EMBD * 2);
  __bf16* WTz    = (__bf16*)take((size_t)EMBD * VOCABD * 2);

  // ---- weight prep: transpose + f32->bf16 (tiny) ----
  auto tconv = [&](const float* in, __bf16* o, int R, int C) {
    tconv_kernel<<<(R * C + 255) / 256, 256, 0, stream>>>(in, o, R, C);
  };
  tconv(W_src1, WTsrc1, HIDD, EMBD);
  tconv(W_src3, WTsrc3, HIDD, EMBD);
  tconv(W_h1, WTh1, EMBD, EMBD);
  tconv(W_h3, WTh3, EMBD, EMBD);
  for (int t = 0; t < NSLT; t++) {
    tconv(W_rel1 + (size_t)t * EMBD * EMBD, WTrel1 + (size_t)t * EMBD * EMBD, EMBD, EMBD);
    tconv(W_rel3 + (size_t)t * EMBD * EMBD, WTrel3 + (size_t)t * EMBD * EMBD, EMBD, EMBD);
  }
  tconv(Wz, WTz, EMBD, VOCABD);

  // ---- embedding: h = sum_l embed_table[tgt_x] ----
  embed_kernel<<<NN, 256, 0, stream>>>(tgt_x, emtab, hA);

  auto gemm = [&](const float* A, const __bf16* BT, float* C, float* OUT,
                  int M, int N, int K, int ep) {
    const int strips = (M / 16) * (N / 64);
    gemm_bf16_kernel<<<(strips + 7) / 8, 256, 0, stream>>>(A, BT, C, OUT, M, N, K, ep);
  };

  // ---- three GCN layers: layer1, layer1, layer3 (ping-pong h/acc) ----
  float* h_cur = hA;
  float* acc   = hB;
  for (int layer = 0; layer < 3; layer++) {
    const __bf16* WTsrc = (layer < 2) ? WTsrc1 : WTsrc3;
    const __bf16* WTh   = (layer < 2) ? WTh1   : WTh3;
    const __bf16* WTrel = (layer < 2) ? WTrel1 : WTrel3;

    // k = x @ W_src   [1024 x 256]
    gemm(x, WTsrc, kf32, kf32, MSRC, EMBD, HIDD, 0);
    kprep_kernel<<<(MSRC * EMBD) / 256, 256, 0, stream>>>(kf32, kbf, kTbf);

    // acc = ctx = softmax(h k^T / 16) k   (fused, WMMA both GEMMs)
    attn_kernel<<<NN / 16, 256, 0, stream>>>(h_cur, kbf, kTbf, acc);

    // acc += sum_t (scatter_t(h)) @ W_rel[t]
    for (int t = 0; t < NSLT; t++) {
      zero_kernel<<<((NN * EMBD / 4) + 255) / 256, 256, 0, stream>>>(
          (float4*)S_buf, NN * EMBD / 4);
      scatter_kernel<<<(EE + 7) / 8, 256, 0, stream>>>(h_cur, eidx, etype, S_buf, t);
      gemm(S_buf, WTrel + (size_t)t * EMBD * EMBD, acc, acc, NN, EMBD, EMBD, 1);
    }

    // acc = relu(acc + h @ W_h)   (fused epilogue, in-place on acc)
    gemm(h_cur, WTh, acc, acc, NN, EMBD, EMBD, 2);

    // ping-pong: relu output becomes next layer's h
    float* tmp = h_cur; h_cur = acc; acc = tmp;
  }

  // ---- node head: logits -> d_out[0 : N*VOCAB], in-place log_softmax ----
  gemm(h_cur, WTz, out, out, NN, VOCABD, EMBD, 0);
  node_lsm_kernel<<<(NN + 7) / 8, 256, 0, stream>>>(out, bz);

  // ---- edge head -> d_out[N*VOCAB : ] ----
  edge_head_kernel<<<(EE + 7) / 8, 256, 0, stream>>>(
      h_cur, eidx, Wg, bg, out + (size_t)NN * VOCABD);
}
